// PoolingLayer_22273700397665
// MI455X (gfx1250) — compile-verified, matrix operand order checked by reading, not run on hardware
//
#include <hip/hip_runtime.h>
#include <hip/hip_bf16.h>

typedef __attribute__((ext_vector_type(16))) __bf16 v16bf;
typedef __attribute__((ext_vector_type(8)))  float  v8f;
typedef int v4i __attribute__((vector_size(16)));   // matches builtin param type

#define BB    2048
#define SS    200
#define DSEQ  256
#define DITEM 64
#define AA    128
#define STILE 16
#define NT    13       // 13 tiles of 16 cover S=200
#define ROWP  260      // padded LDS row stride (floats): rotates banks by 4/row

#if defined(__has_builtin)
#if __has_builtin(__builtin_amdgcn_global_load_async_to_lds_b128) && \
    __has_builtin(__builtin_amdgcn_s_wait_asynccnt)
#define USE_ASYNC 1
#endif
#endif
#ifndef USE_ASYNC
#define USE_ASYNC 0
#endif

#define AS1 __attribute__((address_space(1)))
#define AS3 __attribute__((address_space(3)))

#if USE_ASYNC
__device__ __forceinline__ AS1 v4i* to_global_v4(const void* p) {
  return (AS1 v4i*)(const_cast<void*>(p));    // addrspacecast generic->global
}
__device__ __forceinline__ AS3 v4i* to_lds_v4(void* p) {
  return (AS3 v4i*)p;                         // addrspacecast generic->LDS
}
#endif

// ---- prep: fp32 -> bf16 weight conversion (runs once per launch, tiny) ----
__global__ void cdna5_convert_weights(const float* __restrict__ wk,
                                      const float* __restrict__ wv,
                                      __bf16* __restrict__ wkb,
                                      __bf16* __restrict__ wvb) {
  int i = blockIdx.x * blockDim.x + threadIdx.x;
  if (i < AA * DSEQ) {
    wkb[i] = (__bf16)wk[i];
    wvb[i] = (__bf16)wv[i];
  }
}

// ---- fused: K/V projection (WMMA bf16, async double-buffered staging)
//      + bilinear score + online softmax + attn*V + FFN; 1 block / batch ----
__global__ __launch_bounds__(256) void cdna5_fused_attn(
    const float* __restrict__ X,      // [B,S,DSEQ]
    const int*   __restrict__ mask,   // [B,S] 1 = pad
    const float* __restrict__ item,   // [B,DITEM]
    const float* __restrict__ Wq,     // [AA,DITEM]
    const float* __restrict__ Wker,   // [AA,AA]
    const __bf16* __restrict__ wkb,   // [AA,DSEQ] bf16
    const __bf16* __restrict__ wvb,   // [AA,DSEQ] bf16
    const float* __restrict__ ffnW,   // [DSEQ,AA]
    const float* __restrict__ ffnB,   // [DSEQ]
    const float* __restrict__ preluA, // scalar
    float* __restrict__ outFfn,       // [B,DSEQ]
    float* __restrict__ outAttn) {    // [B,S]
  __shared__ float inTile[2][STILE * ROWP]; // 2 x 16.25 KB staged activations
  __shared__ float Qs[AA];
  __shared__ float Qts[AA];
  __shared__ float scoreAll[NT * STILE];
  __shared__ float scoreTile[16];
  __shared__ float pTile[16];
  __shared__ float softState[3];            // m, l, c
  __shared__ float outLds[AA];

  const int b    = blockIdx.x;
  const int t    = threadIdx.x;
  const int lane = t & 31;               // wave32
  const int w    = t >> 5;               // 8 waves
  const int half = lane >> 4;
  const int ln16 = lane & 15;
  const float pa = preluA[0];

  const int stageRow = t >> 4;           // 16 threads per row
  const int stageCol = (t & 15) * 16;    // 16 floats each

#if USE_ASYNC
  { // prologue: kick off tile 0 (all rows valid) into buffer 0
    const float* g = X + ((size_t)b * SS + stageRow) * DSEQ + stageCol;
    float* l = &inTile[0][stageRow * ROWP + stageCol];
    #pragma unroll
    for (int j = 0; j < 4; ++j)
      __builtin_amdgcn_global_load_async_to_lds_b128(to_global_v4(g + 4 * j),
                                                     to_lds_v4(l + 4 * j), 0, 0);
  }
#endif

  // ---- Q = prelu(Wq @ item[b]); Qt = Wker @ Q (overlaps with async tile 0) ----
  if (t < AA) {
    float s = 0.f;
    const float* wq = Wq + t * DITEM;
    const float* it = item + (size_t)b * DITEM;
    #pragma unroll 8
    for (int d = 0; d < DITEM; ++d) s += wq[d] * it[d];
    Qs[t] = (s >= 0.f) ? s : pa * s;
  }
  if (t == 0) { softState[0] = -3.0e38f; softState[1] = 0.f; softState[2] = 0.f; }
  __syncthreads();
  if (t < AA) {
    float s = 0.f;
    const float* wr = Wker + t * AA;
    #pragma unroll 8
    for (int c = 0; c < AA; ++c) s += wr[c] * Qs[c];
    Qts[t] = s;
  }
  __syncthreads();

  const float qtLane = Qts[16 * w + ln16];   // Qt for this lane's 'a' column
  float accA = 0.f;                          // online attn*V acc (a = 16w+ln16)

  for (int it = 0; it < NT; ++it) {
    const int s0  = it * STILE;
    const int cur = it & 1;
    const int nxt = cur ^ 1;

#if USE_ASYNC
    // (A) issue async copy of tile it+1 into the other buffer (or zero-pad)
    bool issuedNext = false;                 // wave-uniform
    if (it + 1 < NT) {
      const int sN = s0 + STILE + stageRow;
      float* l = &inTile[nxt][stageRow * ROWP + stageCol];
      if (sN < SS) {
        const float* g = X + ((size_t)b * SS + sN) * DSEQ + stageCol;
        #pragma unroll
        for (int j = 0; j < 4; ++j)
          __builtin_amdgcn_global_load_async_to_lds_b128(to_global_v4(g + 4 * j),
                                                         to_lds_v4(l + 4 * j), 0, 0);
        issuedNext = true;
      } else {                               // rows past S: zero via plain LDS
        float4 z = make_float4(0, 0, 0, 0);
        float4* d = (float4*)l;
        d[0] = z; d[1] = z; d[2] = z; d[3] = z;
      }
    }
    if (t < 16) scoreTile[t] = 0.f;
    // (W) drain this wave's copies for tile `it` (in-order completion),
    //     leaving the just-issued next-tile copies in flight
    if (issuedNext) __builtin_amdgcn_s_wait_asynccnt(4);
    else            __builtin_amdgcn_s_wait_asynccnt(0);
    __syncthreads();                         // tile `it` visible to all waves
    const float* tileBase = &inTile[cur][0];
#else
    __syncthreads();                         // B1: prev tile fully consumed
    if (t < 16) scoreTile[t] = 0.f;
    { // synchronous staging fallback
      int s = s0 + stageRow;
      float4 v0, v1, v2, v3;
      if (s < SS) {
        const float4* src =
            (const float4*)(X + ((size_t)b * SS + s) * DSEQ + stageCol);
        v0 = src[0]; v1 = src[1]; v2 = src[2]; v3 = src[3];
        if (s + STILE < SS)
          __builtin_prefetch(X + ((size_t)b * SS + s + STILE) * DSEQ + stageCol, 0, 1);
      } else {
        v0 = make_float4(0, 0, 0, 0); v1 = v0; v2 = v0; v3 = v0;
      }
      float4* dst = (float4*)(&inTile[0][0] + stageRow * ROWP + stageCol);
      dst[0] = v0; dst[1] = v1; dst[2] = v2; dst[3] = v3;
    }
    __syncthreads();                         // B2: tile visible
    const float* tileBase = &inTile[0][0];
#endif

    // ---- K/V projection: [16s x 256] x [256 x 16a] per wave, bf16 WMMA ----
    v8f kacc = {0, 0, 0, 0, 0, 0, 0, 0};
    v8f vacc = {0, 0, 0, 0, 0, 0, 0, 0};
    const int aCol = 16 * w + ln16;
    #pragma unroll
    for (int k0 = 0; k0 < DSEQ; k0 += 32) {
      // A fragment: row M=ln16, 16 consecutive k (half selects k-range)
      const float* ap = tileBase + ln16 * ROWP + k0 + half * 16;
      v16bf afrag;
      #pragma unroll
      for (int j = 0; j < 16; ++j) afrag[j] = (__bf16)ap[j];
      // B fragments: column N=aCol, 16 consecutive k of W^T (= row-major W);
      // loop-invariant -> compiler keeps all 16 fragments resident in VGPRs
      const v16bf bk = *(const v16bf*)(wkb + (size_t)aCol * DSEQ + k0 + half * 16);
      const v16bf bv = *(const v16bf*)(wvb + (size_t)aCol * DSEQ + k0 + half * 16);
      kacc = __builtin_amdgcn_wmma_f32_16x16x32_bf16(false, afrag, false, bk,
                                                     (short)0, kacc, false, false);
      vacc = __builtin_amdgcn_wmma_f32_16x16x32_bf16(false, afrag, false, bv,
                                                     (short)0, vacc, false, false);
    }

    // ---- prelu + bilinear score partials: score[s] += Kp[s,a]*Qt[a] ----
    #pragma unroll
    for (int r = 0; r < 8; ++r) {
      float kx = kacc[r]; kx = (kx >= 0.f) ? kx : pa * kx;
      float vx = vacc[r]; vacc[r] = (vx >= 0.f) ? vx : pa * vx;
      float sp = kx * qtLane;                // reduce over 16 'a' lanes
      sp += __shfl_xor(sp, 8, 16);
      sp += __shfl_xor(sp, 4, 16);
      sp += __shfl_xor(sp, 2, 16);
      sp += __shfl_xor(sp, 1, 16);
      if (ln16 == 0) atomicAdd(&scoreTile[r + 8 * half], sp);  // ds_add_f32
    }
    __syncthreads();                         // B3: scoreTile complete

    // ---- online softmax update (thread 0) ----
    if (t == 0) {
      const float inv = 0.08838834764831845f; // 1/sqrt(128)
      float m = softState[0];
      float tmax = m;
      float sc[16];
      #pragma unroll
      for (int i = 0; i < 16; ++i) {
        int s = s0 + i;
        float v = -3.0e38f;
        if (s < SS) {
          v = (mask[(size_t)b * SS + s] != 0) ? -1.0e9f : scoreTile[i] * inv;
          scoreAll[s] = v;
          if (v > tmax) tmax = v;
        }
        sc[i] = v;
      }
      float c = __expf(m - tmax);
      float sum = 0.f;
      #pragma unroll
      for (int i = 0; i < 16; ++i) {
        float p = (s0 + i < SS) ? __expf(sc[i] - tmax) : 0.f;
        pTile[i] = p; sum += p;
      }
      softState[0] = tmax;
      softState[1] = softState[1] * c + sum;
      softState[2] = c;
    }
    __syncthreads();                         // B4: p, c ready

    // ---- acc[a] = acc[a]*c + sum_s p[s]*Vp[s,a] ----
    {
      float c = softState[2];
      float tsum = 0.f;
      #pragma unroll
      for (int r = 0; r < 8; ++r) tsum += pTile[r + 8 * half] * vacc[r];
      tsum += __shfl_xor(tsum, 16, 32);      // combine s-halves (same a)
      accA = accA * c + tsum;
    }
  }

  const float lFin = softState[1];
  const float mFin = softState[0];
  if (lane < 16) outLds[16 * w + lane] = accA / lFin;
  __syncthreads();

  // ---- FFN: o = t in [0,256) ----
  {
    float s = ffnB[t];
    const float* wr = ffnW + (size_t)t * AA;
    #pragma unroll 8
    for (int a = 0; a < AA; ++a) s += wr[a] * outLds[a];
    s = (s >= 0.f) ? s : pa * s;
    outFfn[(size_t)b * DSEQ + t] = s;
  }
  // ---- attn output ----
  if (t < SS) outAttn[(size_t)b * SS + t] = __expf(scoreAll[t] - mFin) / lFin;
}

extern "C" void kernel_launch(void* const* d_in, const int* in_sizes, int n_in,
                              void* d_out, int out_size, void* d_ws, size_t ws_size,
                              hipStream_t stream) {
  const float* X    = (const float*)d_in[0];
  const int*   mask = (const int*)  d_in[1];
  const float* item = (const float*)d_in[2];
  const float* Wq   = (const float*)d_in[3];
  const float* Wk   = (const float*)d_in[4];
  const float* Wv   = (const float*)d_in[5];
  const float* Wker = (const float*)d_in[6];
  const float* ffnW = (const float*)d_in[7];
  const float* ffnB = (const float*)d_in[8];
  const float* pa   = (const float*)d_in[9];

  __bf16* wkb = (__bf16*)d_ws;              // 64 KB
  __bf16* wvb = wkb + AA * DSEQ;            // 64 KB

  float* out     = (float*)d_out;
  float* outAttn = out + (size_t)BB * DSEQ;

  cdna5_convert_weights<<<dim3((AA * DSEQ + 255) / 256), dim3(256), 0, stream>>>(
      Wk, Wv, wkb, wvb);
  cdna5_fused_attn<<<dim3(BB), dim3(256), 0, stream>>>(
      X, mask, item, Wq, Wker, wkb, wvb, ffnW, ffnB, pa, out, outAttn);
}
